// JointModel_3006477107446
// MI455X (gfx1250) — matrix operand order repaired and checked by reading, not compiled, not run on hardware
//
#include <hip/hip_runtime.h>

// ---------------------------------------------------------------------------
// Fused LSTM (INPUT=8, HIDDEN=50, OUT=4) for B=2048, T=512 on gfx1250.
// One wave32 owns a 16-row batch tile for all 512 timesteps.
// Per step:  g[16x208]  = [h|x|1][16x64] @ Wc[64x208]  (v_wmma_f32_16x16x32_f16)
//            out[16x4]  = [h|x|1][16x64] @ Wo[64x16]
// Biases ride in the K=58 "ones" column of the augmented matmul.
// Weights live in registers (28 loop-invariant B-fragments); h/x live in LDS
// f16; c lives in registers (25 floats/lane). Gates use hw v_tanh_f32.
// ---------------------------------------------------------------------------

typedef __attribute__((ext_vector_type(16))) _Float16 v16h;
typedef __attribute__((ext_vector_type(8)))  _Float16 v8h;
typedef __attribute__((ext_vector_type(2)))  _Float16 v2h;
typedef __attribute__((ext_vector_type(8)))  float    v8f;
typedef __attribute__((ext_vector_type(4)))  float    v4f;

#define WMMA_F16(A, B, C) \
  __builtin_amdgcn_wmma_f32_16x16x32_f16(false, (A), false, (B), (short)0, (C), false, false)

constexpr int BATCH  = 2048;
constexpr int TSTEPS = 512;
constexpr int IN   = 8;
constexpr int H    = 50;
constexpr int G4   = 200;   // 4*H
constexpr int OUTD = 4;
constexpr int NT   = 13;    // N-tiles of 16 covering 208 (>=200)
constexpr int BCOL = H + IN;  // 58: the "ones" column carrying biases
constexpr int WCS  = 72;    // Wc row stride in halves (bank-spread pad, 16B-mult)
constexpr int HXS  = 72;    // hx row stride in halves
constexpr int GSTR = 212;   // g row stride in floats (cols 208..211 = out staging)
constexpr int WAVES = 4;    // waves per block, each independent

// A-matrix fragment (16x32 f16): lane m=L%16; halves 0..7 <- K=k0+(L/16)*8+0..7,
// halves 8..15 <- K=k0+16+(L/16)*8+0..7.
__device__ __forceinline__ v16h ldA(const _Float16* row, int k0, int hi) {
  union { v16h v; v8h h[2]; } u;
  u.h[0] = *(const v8h*)(row + k0 + hi * 8);
  u.h[1] = *(const v8h*)(row + k0 + 16 + hi * 8);
  return u.v;
}

// B-matrix fragment (32x16 f16): lane n=L%16; halves 0..7 <- K=k0+(L/16)*8+0..7,
// halves 8..15 <- K=k0+16+(L/16)*8+0..7.
__device__ __forceinline__ v16h ldB(const _Float16* row, int k0, int hi) {
  union { v16h v; v8h h[2]; } u;
  const _Float16* p = row + k0 + hi * 8;
  u.h[0] = *(const v8h*)(p);
  u.h[1] = *(const v8h*)(p + 16);
  return u.v;
}

__device__ __forceinline__ float tanh_fast(float v) {
#if __has_builtin(__builtin_amdgcn_tanhf)
  return __builtin_amdgcn_tanhf(v);          // gfx1250 v_tanh_f32
#else
  v = fmaxf(-15.f, fminf(15.f, v));
  float e = __expf(-2.f * v);
  return (1.f - e) * __builtin_amdgcn_rcpf(1.f + e);
#endif
}

__device__ __forceinline__ float sig_fast(float v) {
  // sigmoid(x) = 0.5*tanh(0.5x) + 0.5
  return __builtin_fmaf(0.5f, tanh_fast(0.5f * v), 0.5f);
}

__global__ __launch_bounds__(128, 1) void lstm_wmma_kernel(
    const float* __restrict__ x,     const float* __restrict__ Wih,
    const float* __restrict__ Whh,   const float* __restrict__ bih,
    const float* __restrict__ bhh,   const float* __restrict__ Wout,
    const float* __restrict__ boutg, float* __restrict__ out)
{
  __shared__ __align__(16) _Float16 sWc[NT * 16 * WCS];     // [208][72] f16
  __shared__ __align__(16) _Float16 sWo[16 * WCS];          // [16][72]  f16
  __shared__ __align__(16) _Float16 sHX[WAVES * 16 * HXS];  // per-wave [16][72] f16
  __shared__ __align__(16) float    sG[WAVES * 16 * GSTR];  // per-wave [16][212] f32

  const int tid = threadIdx.x;

  // ---- one-time staging of weights (+ biases in col 58) into LDS ----
  for (int idx = tid; idx < NT * 16 * 64; idx += 128) {
    int n = idx >> 6, k = idx & 63;
    float v = 0.f;
    if (n < G4) {
      if (k < H)            v = Whh[n * H + k];
      else if (k < H + IN)  v = Wih[n * IN + (k - H)];
      else if (k == BCOL)   v = bih[n] + bhh[n];
    }
    sWc[n * WCS + k] = (_Float16)v;
  }
  for (int idx = tid; idx < 16 * 64; idx += 128) {
    int n = idx >> 6, k = idx & 63;
    float v = 0.f;
    if (n < OUTD) {
      if (k < H)          v = Wout[n * H + k];
      else if (k == BCOL) v = boutg[n];
    }
    sWo[n * WCS + k] = (_Float16)v;
  }
  // h = 0, x slots = 0, "ones" column = 1
  for (int idx = tid; idx < WAVES * 16 * HXS; idx += 128)
    sHX[idx] = ((idx % HXS) == BCOL) ? (_Float16)1.f : (_Float16)0.f;
  __syncthreads();

  const int w    = tid >> 5;
  const int lane = tid & 31;
  const int lm   = lane & 15;
  const int hi   = lane >> 4;
  _Float16* hx = &sHX[w * 16 * HXS];
  float*    gw = &sG[w * 16 * GSTR];
  const int b0 = (blockIdx.x * WAVES + w) * 16;

  // ---- loop-invariant B fragments into registers ----
  v16h Bf[2 * NT];
#pragma unroll
  for (int nt = 0; nt < NT; ++nt) {
#pragma unroll
    for (int q = 0; q < 2; ++q)
      Bf[nt * 2 + q] = ldB(&sWc[(nt * 16 + lm) * WCS], q * 32, hi);
  }
  v16h Bo0 = ldB(&sWo[lm * WCS], 0, hi);
  v16h Bo1 = ldB(&sWo[lm * WCS], 32, hi);

  float c[25];
#pragma unroll
  for (int i = 0; i < 25; ++i) c[i] = 0.f;

  // x staging: lane -> (row m = lane/2, 4 floats at col (lane&1)*4)
  const int xm = lane >> 1;
  const int xs = (lane & 1) * 4;
  const float* xrow = x + (size_t)(b0 + xm) * TSTEPS * IN + xs;

  for (int t = 0; t < TSTEPS; ++t) {
    // stage x_t (f32 -> f16) into hx[., 50..57]
    v4f xv = *(const v4f*)(xrow + (size_t)t * IN);
    v2h p0; p0[0] = (_Float16)xv[0]; p0[1] = (_Float16)xv[1];
    v2h p1; p1[0] = (_Float16)xv[2]; p1[1] = (_Float16)xv[3];
    *(v2h*)&hx[xm * HXS + H + xs]     = p0;
    *(v2h*)&hx[xm * HXS + H + xs + 2] = p1;
    if (t + 1 < TSTEPS) __builtin_prefetch(xrow + (size_t)(t + 1) * IN, 0, 1);

    // A fragments of [h | x_t | 1] (K = 0..63)
    v16h A0 = ldA(&hx[lm * HXS], 0, hi);
    v16h A1 = ldA(&hx[lm * HXS], 32, hi);

    // gates (bias included): g[16 x 208] = A @ Wc
#pragma unroll
    for (int nt = 0; nt < NT; ++nt) {
      v8f d = {};
      d = WMMA_F16(A0, Bf[2 * nt],     d);
      d = WMMA_F16(A1, Bf[2 * nt + 1], d);
      const int n = nt * 16 + lm;
#pragma unroll
      for (int r = 0; r < 8; ++r) gw[(r + 8 * hi) * GSTR + n] = d[r];
    }

    // fused output projection (b_out included): out_t[16 x 4] = [h|x|1] @ Wo
    v8f dq = {};
    dq = WMMA_F16(A0, Bo0, dq);
    dq = WMMA_F16(A1, Bo1, dq);
    if (lm < OUTD) {
#pragma unroll
      for (int r = 0; r < 8; ++r) gw[(r + 8 * hi) * GSTR + 208 + lm] = dq[r];
    }
    // coalesced 16B store per batch row
    if (lane < 16) {
      v4f o4 = *(const v4f*)&gw[lane * GSTR + 208];
      *(v4f*)&out[((size_t)(b0 + lane) * TSTEPS + t) * OUTD] = o4;
    }

    // elementwise gate stage: 800 (m,j) pairs over 32 lanes = 25/lane
#pragma unroll
    for (int it = 0; it < 25; ++it) {
      const unsigned e = (unsigned)(it * 32 + lane);
      const unsigned m = e / 50u;
      const unsigned j = e - m * 50u;
      const float* gr = &gw[m * GSTR + j];
      float gi = gr[0];
      float gf = gr[50];
      float gg = gr[100];
      float go = gr[150];
      float iv = sig_fast(gi), fv = sig_fast(gf), ov = sig_fast(go);
      float tg = tanh_fast(gg);
      float cn = __builtin_fmaf(fv, c[it], iv * tg);
      c[it] = cn;
      float hv = ov * tanh_fast(cn);
      hx[m * HXS + j] = (_Float16)hv;
    }
  }
}

extern "C" void kernel_launch(void* const* d_in, const int* in_sizes, int n_in,
                              void* d_out, int out_size, void* d_ws, size_t ws_size,
                              hipStream_t stream) {
  (void)in_sizes; (void)n_in; (void)out_size; (void)d_ws; (void)ws_size;
  const float* x    = (const float*)d_in[0];
  const float* Wih  = (const float*)d_in[1];
  const float* Whh  = (const float*)d_in[2];
  const float* bih  = (const float*)d_in[3];
  const float* bhh  = (const float*)d_in[4];
  const float* Wout = (const float*)d_in[5];
  const float* bout = (const float*)d_in[6];
  float* out = (float*)d_out;

  dim3 grid(BATCH / (16 * WAVES));  // 32 blocks
  dim3 block(32 * WAVES);           // 128 threads = 4 wave32
  lstm_wmma_kernel<<<grid, block, 0, stream>>>(x, Wih, Whh, bih, bhh, Wout, bout, out);
}
